// LSTM_MNIST_11708080849331
// MI455X (gfx1250) — compile-verified, matrix operand order checked by reading, not run on hardware
//
#include <hip/hip_runtime.h>

#define BATCH   256
#define SEQ     64
#define INPUT   512
#define HIDDEN  1024
#define CLASSES 1000
#define G4H     (4 * HIDDEN)

typedef __bf16 bf16;
typedef __attribute__((ext_vector_type(16))) __bf16 v16bf;
typedef __attribute__((ext_vector_type(8)))  __bf16 v8bf;
typedef __attribute__((ext_vector_type(8)))  float  v8f;

// ---------------------------------------------------------------------------
// 16x32 bf16 fragment loader (CDNA5 ISA §7.12.2 A-matrix layout; B fragments
// load the same pattern from the transposed [N][K] weight layout):
//   lanes 0-15 : row = lane,    K {0..7} then {16..23}
//   lanes 16-31: row = lane-16, K {8..15} then {24..31}
// => two contiguous 16-byte global_load_b128 per lane.
// ---------------------------------------------------------------------------
__device__ __forceinline__ v16bf load_frag16x32(const bf16* __restrict__ base,
                                                long long rowStride,
                                                int lane, int kBase) {
  const int half = lane >> 4;
  const int r    = lane & 15;
  const bf16* p  = base + (long long)r * rowStride + kBase + half * 8;
  v8bf lo = *(const v8bf*)(p);
  v8bf hi = *(const v8bf*)(p + 16);
  v16bf out;
#pragma unroll
  for (int i = 0; i < 8; ++i) { out[i] = lo[i]; out[i + 8] = hi[i]; }
  return out;
}

__device__ __forceinline__ void wmma4(v8f acc[4], const v16bf a[4], v16bf b) {
#pragma unroll
  for (int i = 0; i < 4; ++i)
    acc[i] = __builtin_amdgcn_wmma_f32_16x16x32_bf16(false, a[i], false, b,
                                                     (short)0, acc[i], false, false);
}

__device__ __forceinline__ void load_set4(v16bf a[4], v16bf& b,
                                          const bf16* __restrict__ aBase, long long aStride,
                                          const bf16* __restrict__ bBase, long long bStride,
                                          int lane, int k) {
  b = load_frag16x32(bBase, bStride, lane, k);
#pragma unroll
  for (int i = 0; i < 4; ++i)
    a[i] = load_frag16x32(aBase + (long long)(i * 16) * aStride, aStride, lane, k);
}

// ---------------------------------------------------------------------------
// Fully-unrolled K-loop (K compile-time: 512 or 1024), 4 m-tiles per wave
// sharing one B fragment.  Ping-pong register sets + full unroll => the
// double-buffer rotation costs zero v_mov copies and no backedge hazard NOPs;
// the scheduler hoists loads as far ahead as registers allow.
// ---------------------------------------------------------------------------
template <int K>
__device__ __forceinline__ void accum_gemm4(v8f acc[4],
                                            const bf16* __restrict__ aBase,  // 64 rows
                                            long long aStride,
                                            const bf16* __restrict__ bBase,
                                            long long bStride,
                                            int lane) {
  constexpr int S = K >> 5;                  // #k-steps, even (16 or 32)
  v16bf a0[4], a1[4], b0, b1;
  load_set4(a0, b0, aBase, aStride, bBase, bStride, lane, 0);
#pragma unroll
  for (int s = 1; s + 1 < S; s += 2) {
    __builtin_prefetch(bBase + (long long)(lane & 15) * bStride + (s + 1) * 32, 0, 1);
    load_set4(a1, b1, aBase, aStride, bBase, bStride, lane, s * 32);
    wmma4(acc, a0, b0);
    load_set4(a0, b0, aBase, aStride, bBase, bStride, lane, (s + 1) * 32);
    wmma4(acc, a1, b1);
  }
  load_set4(a1, b1, aBase, aStride, bBase, bStride, lane, (S - 1) * 32);
  wmma4(acc, a0, b0);
  wmma4(acc, a1, b1);
}

// Single-tile fully-unrolled variant (projection / bulk-GEMM kernels).
template <int K>
__device__ __forceinline__ v8f accum_gemm(v8f acc,
                                          const bf16* __restrict__ aBase, long long aStride,
                                          const bf16* __restrict__ bBase, long long bStride,
                                          int lane) {
  constexpr int S = K >> 5;
  v16bf a0, a1, b0, b1;
  a0 = load_frag16x32(aBase, aStride, lane, 0);
  b0 = load_frag16x32(bBase, bStride, lane, 0);
#pragma unroll
  for (int s = 1; s + 1 < S; s += 2) {
    a1 = load_frag16x32(aBase, aStride, lane, s * 32);
    b1 = load_frag16x32(bBase, bStride, lane, s * 32);
    acc = __builtin_amdgcn_wmma_f32_16x16x32_bf16(false, a0, false, b0, (short)0, acc, false, false);
    a0 = load_frag16x32(aBase, aStride, lane, (s + 1) * 32);
    b0 = load_frag16x32(bBase, bStride, lane, (s + 1) * 32);
    acc = __builtin_amdgcn_wmma_f32_16x16x32_bf16(false, a1, false, b1, (short)0, acc, false, false);
  }
  a1 = load_frag16x32(aBase, aStride, lane, (S - 1) * 32);
  b1 = load_frag16x32(bBase, bStride, lane, (S - 1) * 32);
  acc = __builtin_amdgcn_wmma_f32_16x16x32_bf16(false, a0, false, b0, (short)0, acc, false, false);
  return __builtin_amdgcn_wmma_f32_16x16x32_bf16(false, a1, false, b1, (short)0, acc, false, false);
}

__device__ __forceinline__ float sigmoidf_dev(float x) {
  return 1.0f / (1.0f + __expf(-x));
}

// ---------------------------------------------------------------------------
// Fused LSTM cell.  Block = 128 threads = 4 waves; wave g computes gate g's
// z-tiles for a 64(batch) x 16(hidden) strip (4 WMMA tiles), tiles meet in
// LDS, then the block cooperatively applies the nonlinearity.  If zPre !=
// null, the x@W contribution was precomputed and seeds the accumulators.
// KX = K-extent of the Xa operand (compile-time).
// ---------------------------------------------------------------------------
template <int KX>
__global__ void __launch_bounds__(128)
lstm_cell_wmma(const float* __restrict__ zPre,   // [BATCH][4H] for this t, or null
               const bf16* __restrict__ Xa, long long xRowStride,
               const bf16* __restrict__ Wt,      // [4H][KX]  (transposed)
               const bf16* __restrict__ Ha,      // [BATCH][HIDDEN]
               const bf16* __restrict__ Ut,      // [4H][HIDDEN] (transposed)
               const float* __restrict__ bias,   // [4H]
               float* __restrict__ cState,       // [BATCH][HIDDEN], in/out
               bf16* __restrict__ hOut)          // [BATCH][HIDDEN]
{
  __shared__ float zbuf[4][4][256];              // [gate][m-tile][16x16]

  const int lane   = threadIdx.x & 31;
  const int gate   = threadIdx.x >> 5;           // wave id == gate index
  const int nBase  = blockIdx.x * 16;            // hidden-column tile
  const int mBase0 = blockIdx.y * 64;            // 64-row batch strip
  const int nl     = lane & 15;
  const int half   = lane >> 4;

  v8f acc[4] = {v8f{}, v8f{}, v8f{}, v8f{}};

  if (zPre != nullptr) {
    // seed from precomputed x @ W (bias added later)
#pragma unroll
    for (int i = 0; i < 4; ++i) {
      const float* zp = zPre + (long long)(mBase0 + i * 16 + half * 8) * G4H
                             + (long long)gate * HIDDEN + nBase + nl;
#pragma unroll
      for (int j = 0; j < 8; ++j) acc[i][j] = zp[(long long)j * G4H];
    }
  } else {
    accum_gemm4<KX>(acc, Xa + (long long)mBase0 * xRowStride, xRowStride,
                    Wt + (long long)(gate * HIDDEN + nBase) * KX, KX, lane);
  }
  accum_gemm4<HIDDEN>(acc, Ha + (long long)mBase0 * HIDDEN, HIDDEN,
                      Ut + (long long)(gate * HIDDEN + nBase) * HIDDEN, HIDDEN,
                      lane);

  // publish gate tiles: element (m_local, n_local) at m_local*16 + n_local
#pragma unroll
  for (int i = 0; i < 4; ++i)
#pragma unroll
    for (int j = 0; j < 8; ++j)
      zbuf[gate][i][(half * 8 + j) * 16 + nl] = acc[i][j];
  __syncthreads();

  // cooperative cell nonlinearity: 1024 elements / 128 threads = 8 each
  for (int e = threadIdx.x; e < 1024; e += 128) {
    const int mt  = e >> 8;          // which 16x16 m-tile
    const int r   = e & 255;         // element within tile
    const int ml  = r >> 4;
    const int nc  = r & 15;
    const int col = nBase + nc;
    const long long idx = (long long)(mBase0 + mt * 16 + ml) * HIDDEN + col;
    const float zi = zbuf[0][mt][r] + bias[0 * HIDDEN + col];
    const float zf = zbuf[1][mt][r] + bias[1 * HIDDEN + col];
    const float zg = zbuf[2][mt][r] + bias[2 * HIDDEN + col];
    const float zo = zbuf[3][mt][r] + bias[3 * HIDDEN + col];
    const float cOld = cState[idx];
    const float cNew = sigmoidf_dev(zf) * cOld + sigmoidf_dev(zi) * tanhf(zg);
    cState[idx] = cNew;
    hOut[idx] = (bf16)(sigmoidf_dev(zo) * tanhf(cNew));
  }
}

// ---------------------------------------------------------------------------
// Bulk GEMM: zx[t][b][:] = x[b][t][:] @ W1t^T  (no bias), f32 out.
// Fully parallel over all 64 timesteps; removes x@W1 from the recurrent
// critical path.  Logical row = t*BATCH + b; 16-row tiles share t.
// ---------------------------------------------------------------------------
__global__ void __launch_bounds__(128)
xw_gemm_wmma(const bf16* __restrict__ xbf,   // [BATCH][SEQ][INPUT]
             const bf16* __restrict__ W1t,   // [4H][INPUT] (transposed)
             float* __restrict__ zx)         // [SEQ][BATCH][4H]
{
  const int lane  = threadIdx.x & 31;
  const int wave  = threadIdx.x >> 5;
  const int nBase = blockIdx.x * 16;                    // 0..4H
  const int row16 = blockIdx.y * 4 + wave;              // tile of 16 rows
  const int t     = row16 >> 4;                         // BATCH/16 tiles per t
  const int bBase = (row16 & 15) * 16;
  const int nl    = lane & 15;
  const int half  = lane >> 4;

  v8f acc = {};
  const bf16* aB = xbf + ((long long)bBase * SEQ + t) * INPUT;
  acc = accum_gemm<INPUT>(acc, aB, (long long)SEQ * INPUT,
                          W1t + (long long)nBase * INPUT, INPUT, lane);

  float* zp = zx + ((long long)t * BATCH + bBase + half * 8) * G4H + nBase + nl;
#pragma unroll
  for (int j = 0; j < 8; ++j) zp[(long long)j * G4H] = acc[j];
}

// ---------------------------------------------------------------------------
// Output projection: out[B][CLASSES] = h2 @ Wo + bo (WoT zero-padded to 1024).
// ---------------------------------------------------------------------------
__global__ void __launch_bounds__(128)
proj_wmma(const bf16* __restrict__ H2,      // [BATCH][HIDDEN]
          const bf16* __restrict__ WoT,     // [1024(pad N)][HIDDEN]
          const float* __restrict__ bo,     // [CLASSES]
          float* __restrict__ out)          // [BATCH][CLASSES]
{
  const int lane  = threadIdx.x & 31;
  const int wave  = threadIdx.x >> 5;
  const int nBase = blockIdx.x * 16;
  const int mBase = (blockIdx.y * 4 + wave) * 16;

  v8f acc = {};
  acc = accum_gemm<HIDDEN>(acc, H2 + (long long)mBase * HIDDEN, HIDDEN,
                           WoT + (long long)nBase * HIDDEN, HIDDEN, lane);

  const int col  = nBase + (lane & 15);
  const int half = lane >> 4;
  if (col < CLASSES) {
    const float bb = bo[col];
#pragma unroll
    for (int j = 0; j < 8; ++j) {
      const int m = mBase + half * 8 + j;
      out[(long long)m * CLASSES + col] = acc[j] + bb;
    }
  }
}

// ---------------------------------------------------------------------------
// Prep kernels
// ---------------------------------------------------------------------------
// out[n][k] = (n < N) ? bf16(in[k][n]) : 0, out shaped [Npad][K]
__global__ void transpose_cast_kernel(const float* __restrict__ in,
                                      bf16* __restrict__ out,
                                      int K, int N, int Npad) {
  const long long i = (long long)blockIdx.x * blockDim.x + threadIdx.x;
  const long long total = (long long)Npad * K;
  if (i >= total) return;
  const int k = (int)(i % K);
  const int n = (int)(i / K);
  out[i] = (n < N) ? (bf16)in[(long long)k * N + n] : (bf16)0.0f;
}

__global__ void cast_kernel(const float* __restrict__ in,
                            bf16* __restrict__ out, long long n) {
  const long long i = (long long)blockIdx.x * blockDim.x + threadIdx.x;
  if (i < n) out[i] = (bf16)in[i];
}

// ---------------------------------------------------------------------------
extern "C" void kernel_launch(void* const* d_in, const int* in_sizes, int n_in,
                              void* d_out, int out_size, void* d_ws, size_t ws_size,
                              hipStream_t stream) {
  (void)in_sizes; (void)n_in; (void)out_size;

  const float* x  = (const float*)d_in[0];
  const float* W1 = (const float*)d_in[1];
  const float* U1 = (const float*)d_in[2];
  const float* b1 = (const float*)d_in[3];
  const float* W2 = (const float*)d_in[4];
  const float* U2 = (const float*)d_in[5];
  const float* b2 = (const float*)d_in[6];
  const float* Wo = (const float*)d_in[7];
  const float* bo = (const float*)d_in[8];

  char* ws = (char*)d_ws;
  size_t off = 0;
  auto alloc = [&](size_t bytes) -> void* {
    void* p = ws + off;
    off = (off + bytes + 255) & ~(size_t)255;
    return p;
  };

  bf16* W1t = (bf16*)alloc((size_t)G4H * INPUT  * sizeof(bf16)); // [4H][512]
  bf16* U1t = (bf16*)alloc((size_t)G4H * HIDDEN * sizeof(bf16)); // [4H][1024]
  bf16* W2t = (bf16*)alloc((size_t)G4H * HIDDEN * sizeof(bf16));
  bf16* U2t = (bf16*)alloc((size_t)G4H * HIDDEN * sizeof(bf16));
  bf16* WoT = (bf16*)alloc((size_t)HIDDEN * HIDDEN * sizeof(bf16)); // padded N
  bf16* xbf = (bf16*)alloc((size_t)BATCH * SEQ * INPUT * sizeof(bf16));
  bf16* h1bf[2], *h2bf[2];
  h1bf[0] = (bf16*)alloc((size_t)BATCH * HIDDEN * sizeof(bf16));
  h1bf[1] = (bf16*)alloc((size_t)BATCH * HIDDEN * sizeof(bf16));
  h2bf[0] = (bf16*)alloc((size_t)BATCH * HIDDEN * sizeof(bf16));
  h2bf[1] = (bf16*)alloc((size_t)BATCH * HIDDEN * sizeof(bf16));
  float* c1 = (float*)alloc((size_t)BATCH * HIDDEN * sizeof(float));
  float* c2 = (float*)alloc((size_t)BATCH * HIDDEN * sizeof(float));

  // Optional precompute buffer zx = x @ W1 for all timesteps (f32, 268 MB).
  const size_t zxBytes = (size_t)SEQ * BATCH * G4H * sizeof(float);
  const bool usePre = (ws_size >= off + zxBytes);
  float* zx = usePre ? (float*)alloc(zxBytes) : nullptr;

  const int T = 256;
  auto nblk = [](long long n, int t) { return (unsigned)((n + t - 1) / t); };

  // Weight conversion + transpose (bf16, [N][K])
  transpose_cast_kernel<<<nblk((long long)G4H * INPUT, T), T, 0, stream>>>(W1, W1t, INPUT, G4H, G4H);
  transpose_cast_kernel<<<nblk((long long)G4H * HIDDEN, T), T, 0, stream>>>(U1, U1t, HIDDEN, G4H, G4H);
  transpose_cast_kernel<<<nblk((long long)G4H * HIDDEN, T), T, 0, stream>>>(W2, W2t, HIDDEN, G4H, G4H);
  transpose_cast_kernel<<<nblk((long long)G4H * HIDDEN, T), T, 0, stream>>>(U2, U2t, HIDDEN, G4H, G4H);
  transpose_cast_kernel<<<nblk((long long)HIDDEN * HIDDEN, T), T, 0, stream>>>(Wo, WoT, HIDDEN, CLASSES, HIDDEN);
  cast_kernel<<<nblk((long long)BATCH * SEQ * INPUT, T), T, 0, stream>>>(x, xbf, (long long)BATCH * SEQ * INPUT);

  // Zero initial states (stream-capturable async memsets)
  hipMemsetAsync(h1bf[0], 0, (size_t)BATCH * HIDDEN * sizeof(bf16), stream);
  hipMemsetAsync(h2bf[0], 0, (size_t)BATCH * HIDDEN * sizeof(bf16), stream);
  hipMemsetAsync(c1, 0, (size_t)BATCH * HIDDEN * sizeof(float), stream);
  hipMemsetAsync(c2, 0, (size_t)BATCH * HIDDEN * sizeof(float), stream);

  // Bulk x@W1 for all 64 timesteps (off the recurrent critical path)
  if (usePre) {
    xw_gemm_wmma<<<dim3(G4H / 16, (SEQ * BATCH) / 64), dim3(128), 0, stream>>>(xbf, W1t, zx);
  }

  const dim3 grid(HIDDEN / 16, BATCH / 64); // 64 n-tiles x 4 m-strips
  const dim3 blk(128);                      // 4 waves = 4 gates per strip

  for (int t = 0; t < SEQ; ++t) {
    const int rd = t & 1;
    const int wr = rd ^ 1;
    // layer 1: (precomputed x@W1 | x_t) and h1_prev
    lstm_cell_wmma<INPUT><<<grid, blk, 0, stream>>>(
        usePre ? zx + (long long)t * BATCH * G4H : nullptr,
        xbf + (long long)t * INPUT, (long long)SEQ * INPUT,
        W1t, h1bf[rd], U1t, b1, c1, h1bf[wr]);
    // layer 2: h1_new and h2_prev
    lstm_cell_wmma<HIDDEN><<<grid, blk, 0, stream>>>(
        nullptr,
        h1bf[wr], (long long)HIDDEN,
        W2t, h2bf[rd], U2t, b2, c2, h2bf[wr]);
  }

  // SEQ even -> final h2 lives in buffer 0
  proj_wmma<<<dim3(HIDDEN / 16, BATCH / 64), blk, 0, stream>>>(
      h2bf[SEQ & 1], WoT, bo, (float*)d_out);
}
